// ColBERT_List_qa_44521630991138
// MI455X (gfx1250) — matrix operand order repaired and checked by reading, not compiled, not run on hardware
//
#include <hip/hip_runtime.h>
#include <hip/hip_bf16.h>

typedef __attribute__((ext_vector_type(16))) __bf16 v16bf;
typedef __attribute__((ext_vector_type(8)))  float  v8f;

#define NQ   32    // queries
#define MT   32    // query tokens (M)
#define ND   512   // docs
#define NT   180   // doc tokens (N)
#define HH   128   // hidden (K)
#define NTP  192   // padded doc tokens (12 tiles of 16)
#define DSTR 136   // LDS row stride in bf16 elems (128 + 8 pad -> kills bank conflicts)

union Frag {
    v16bf v;
    __bf16 h[16];
    uint4 u4[2];
};

// Raw v_max_num_f32: avoids LLVM's sNaN-canonicalization self-max ops on the
// loop-carried running max.
__device__ __forceinline__ float fmax_raw(float a, float b) {
    float d;
    asm("v_max_num_f32 %0, %1, %2" : "=v"(d) : "v"(a), "v"(b));
    return d;
}

// Native f32 -> bf16 (RNE); pairs should fuse to v_cvt_pk_bf16_f32.
__device__ __forceinline__ unsigned int pack_bf16(float x, float y) {
    union { __bf16 h[2]; unsigned int u; } p;
    p.h[0] = (__bf16)x;
    p.h[1] = (__bf16)y;
    return p.u;
}

__device__ __forceinline__ void load_a_frags(const float* __restrict__ Q,
                                             const int* __restrict__ qmask,
                                             int q, int lrow, int hi16,
                                             Frag a[2][4]) {
    #pragma unroll
    for (int mt = 0; mt < 2; ++mt) {
        const int m = mt * 16 + lrow;
        const float msk = (float)qmask[q * MT + m];
        const float* qr = Q + ((size_t)q * MT + m) * HH;
        const int kb = hi16 * 8;
        #pragma unroll
        for (int ks = 0; ks < 4; ++ks) {
            const float4 r0 = *(const float4*)(qr + ks * 32 + kb);
            const float4 r1 = *(const float4*)(qr + ks * 32 + kb + 4);
            const float4 r2 = *(const float4*)(qr + ks * 32 + 16 + kb);
            const float4 r3 = *(const float4*)(qr + ks * 32 + 16 + kb + 4);
            Frag& f = a[mt][ks];
            f.h[0]  = (__bf16)(r0.x * msk); f.h[1]  = (__bf16)(r0.y * msk);
            f.h[2]  = (__bf16)(r0.z * msk); f.h[3]  = (__bf16)(r0.w * msk);
            f.h[4]  = (__bf16)(r1.x * msk); f.h[5]  = (__bf16)(r1.y * msk);
            f.h[6]  = (__bf16)(r1.z * msk); f.h[7]  = (__bf16)(r1.w * msk);
            f.h[8]  = (__bf16)(r2.x * msk); f.h[9]  = (__bf16)(r2.y * msk);
            f.h[10] = (__bf16)(r2.z * msk); f.h[11] = (__bf16)(r2.w * msk);
            f.h[12] = (__bf16)(r3.x * msk); f.h[13] = (__bf16)(r3.y * msk);
            f.h[14] = (__bf16)(r3.z * msk); f.h[15] = (__bf16)(r3.w * msk);
        }
    }
}

__global__ __launch_bounds__(256)
void colbert_maxsim_kernel(const float* __restrict__ Q,
                           const float* __restrict__ D,
                           const int*   __restrict__ qmask,
                           const int*   __restrict__ dmask,
                           float*       __restrict__ out) {
    __shared__ __align__(16) unsigned short Dlds[NTP * DSTR];  // ~51 KB bf16

    const int d   = blockIdx.x;
    const int tid = threadIdx.x;

    // ---- Phase 1: D[d] (180x128 f32) * d_mask -> bf16 -> LDS (row-major, padded) ----
    for (int e = tid; e < NT * (HH / 4); e += 256) {
        const int n  = e >> 5;          // 32 float4 per row
        const int c4 = e & 31;
        const float m = (float)dmask[d * NT + n];
        const float4 v = *(const float4*)(D + ((size_t)d * NT + n) * HH + c4 * 4);
        const unsigned int lo = pack_bf16(v.x * m, v.y * m);
        const unsigned int hi = pack_bf16(v.z * m, v.w * m);
        *(uint2*)&Dlds[n * DSTR + c4 * 4] = make_uint2(lo, hi);
    }
    // zero-fill pad rows 180..191 (their columns are masked to -inf in the peeled tile)
    for (int e = tid; e < (NTP - NT) * (HH / 4); e += 256) {
        const int n  = NT + (e >> 5);
        const int c4 = e & 31;
        *(uint2*)&Dlds[n * DSTR + c4 * 4] = make_uint2(0u, 0u);
    }
    __syncthreads();

    // ---- Phase 2: per-wave WMMA, 2 queries per wave per pass (B frag reuse x4) ----
    const int lane = tid & 31;
    const int wave = tid >> 5;
    const int lrow = lane & 15;     // row (A) / column (B,C) within tile
    const int hi16 = lane >> 4;     // upper half-wave selector

    for (int qp = 0; qp < 2; ++qp) {
        const int q0 = wave + qp * 16;   // queries q0 and q0+8
        const int q1 = q0 + 8;

        Frag a0[2][4], a1[2][4];
        load_a_frags(Q, qmask, q0, lrow, hi16, a0);
        load_a_frags(Q, qmask, q1, lrow, hi16, a1);

        v8f rmax[4];  // [q0mt0, q0mt1, q1mt0, q1mt1]
        #pragma unroll
        for (int i = 0; i < 4; ++i)
            #pragma unroll
            for (int v = 0; v < 8; ++v) rmax[i][v] = -INFINITY;

        // Full-valid tiles 0..10: plain max merge, no column masking.
        for (int nt = 0; nt < 11; ++nt) {
            v8f acc[4];
            #pragma unroll
            for (int i = 0; i < 4; ++i)
                acc[i] = (v8f){0.f,0.f,0.f,0.f,0.f,0.f,0.f,0.f};
            const int n = nt * 16 + lrow;
            const unsigned short* brow = &Dlds[n * DSTR];
            #pragma unroll
            for (int ks = 0; ks < 4; ++ks) {
                Frag b;
                const int kb = ks * 32 + hi16 * 8;
                b.u4[0] = *(const uint4*)&brow[kb];
                b.u4[1] = *(const uint4*)&brow[kb + 16];
                acc[0] = __builtin_amdgcn_wmma_f32_16x16x32_bf16(
                             false, a0[0][ks].v, false, b.v, (short)0, acc[0], false, false);
                acc[1] = __builtin_amdgcn_wmma_f32_16x16x32_bf16(
                             false, a0[1][ks].v, false, b.v, (short)0, acc[1], false, false);
                acc[2] = __builtin_amdgcn_wmma_f32_16x16x32_bf16(
                             false, a1[0][ks].v, false, b.v, (short)0, acc[2], false, false);
                acc[3] = __builtin_amdgcn_wmma_f32_16x16x32_bf16(
                             false, a1[1][ks].v, false, b.v, (short)0, acc[3], false, false);
            }
            #pragma unroll
            for (int i = 0; i < 4; ++i)
                #pragma unroll
                for (int v = 0; v < 8; ++v)
                    rmax[i][v] = fmax_raw(rmax[i][v], acc[i][v]);
        }

        // Peeled tile 11 (cols 176..191): only cols < 180 (lrow < 4) are valid.
        {
            v8f acc[4];
            #pragma unroll
            for (int i = 0; i < 4; ++i)
                acc[i] = (v8f){0.f,0.f,0.f,0.f,0.f,0.f,0.f,0.f};
            const int n = 11 * 16 + lrow;
            const unsigned short* brow = &Dlds[n * DSTR];
            #pragma unroll
            for (int ks = 0; ks < 4; ++ks) {
                Frag b;
                const int kb = ks * 32 + hi16 * 8;
                b.u4[0] = *(const uint4*)&brow[kb];
                b.u4[1] = *(const uint4*)&brow[kb + 16];
                acc[0] = __builtin_amdgcn_wmma_f32_16x16x32_bf16(
                             false, a0[0][ks].v, false, b.v, (short)0, acc[0], false, false);
                acc[1] = __builtin_amdgcn_wmma_f32_16x16x32_bf16(
                             false, a0[1][ks].v, false, b.v, (short)0, acc[1], false, false);
                acc[2] = __builtin_amdgcn_wmma_f32_16x16x32_bf16(
                             false, a1[0][ks].v, false, b.v, (short)0, acc[2], false, false);
                acc[3] = __builtin_amdgcn_wmma_f32_16x16x32_bf16(
                             false, a1[1][ks].v, false, b.v, (short)0, acc[3], false, false);
            }
            const bool valid = (lrow < (NT - 11 * 16));   // lrow < 4
            #pragma unroll
            for (int i = 0; i < 4; ++i)
                #pragma unroll
                for (int v = 0; v < 8; ++v)
                    rmax[i][v] = fmax_raw(rmax[i][v], valid ? acc[i][v] : -INFINITY);
        }

        // Max over columns: butterfly within each 16-lane half (xor 1,2,4,8)
        #pragma unroll
        for (int off = 1; off <= 8; off <<= 1) {
            #pragma unroll
            for (int i = 0; i < 4; ++i)
                #pragma unroll
                for (int v = 0; v < 8; ++v)
                    rmax[i][v] = fmax_raw(rmax[i][v], __shfl_xor(rmax[i][v], off, 32));
        }
        // Sum over rows: 8 rows per lane-half per M-tile; xor 16 merges halves.
        float s0 = 0.f, s1 = 0.f;
        #pragma unroll
        for (int v = 0; v < 8; ++v) {
            s0 += rmax[0][v] + rmax[1][v];
            s1 += rmax[2][v] + rmax[3][v];
        }
        s0 += __shfl_xor(s0, 16, 32);
        s1 += __shfl_xor(s1, 16, 32);

        if (lane == 0) {
            out[q0 * ND + d] = s0;
            out[q1 * ND + d] = s1;
        }
    }
}

extern "C" void kernel_launch(void* const* d_in, const int* in_sizes, int n_in,
                              void* d_out, int out_size, void* d_ws, size_t ws_size,
                              hipStream_t stream) {
    (void)in_sizes; (void)n_in; (void)out_size; (void)d_ws; (void)ws_size;
    const float* Q  = (const float*)d_in[0];
    const float* D  = (const float*)d_in[1];
    const int* qm   = (const int*)d_in[2];
    const int* dm   = (const int*)d_in[3];
    float* out      = (float*)d_out;
    colbert_maxsim_kernel<<<ND, 256, 0, stream>>>(Q, D, qm, dm, out);
}
